// combinedModel_60808146977353
// MI455X (gfx1250) — matrix operand order repaired and checked by reading, not compiled, not run on hardware
//
#include <hip/hip_runtime.h>
#include <hip/hip_bf16.h>

// ---------------------------------------------------------------------------
// Problem constants
// ---------------------------------------------------------------------------
#define BB 32
#define TT 4096
#define HH 256
#define N1 128           // W1 output dim
#define TILES_TOTAL ((BB * TT) / 16)   // 8192 16-row tiles
#define NWG 256                        // workgroups for the score kernel
#define TILES_PER_WG (TILES_TOTAL / NWG)   // 32
#define WAVES_PER_WG 8

#if defined(__has_builtin)
#if __has_builtin(__builtin_amdgcn_sched_barrier)
#define SCHED_BARRIER(m) __builtin_amdgcn_sched_barrier(m)
#endif
#endif
#ifndef SCHED_BARRIER
#define SCHED_BARRIER(m)
#endif

typedef __bf16 bf16_t;
typedef bf16_t v16bf __attribute__((ext_vector_type(16)));
typedef float  v8f   __attribute__((ext_vector_type(8)));

// ---------------------------------------------------------------------------
// K1: c128[b][n] = b1[n] + sum_k last[b][k] * W1[256+k][n]
// ---------------------------------------------------------------------------
__global__ void c128_kernel(const float* __restrict__ outputs,
                            const float* __restrict__ W1,
                            const float* __restrict__ b1,
                            float* __restrict__ c128) {
    __shared__ float slast[HH];
    const int b = blockIdx.x;
    const int n = threadIdx.x;            // 128 threads
    const float* last = outputs + ((size_t)b * TT + (TT - 1)) * HH;
    slast[n] = last[n];
    slast[n + 128] = last[n + 128];
    __syncthreads();
    float acc = b1[n];
    for (int k = 0; k < HH; ++k)
        acc = fmaf(slast[k], W1[(size_t)(HH + k) * N1 + n], acc);
    c128[b * N1 + n] = acc;
}

// ---------------------------------------------------------------------------
// K2: fused scores kernel.
//   h[16,128] tile = bf16-WMMA( outputs_tile[16,256] , W1[:256,:128] )
//   score[row] = b2 + sum_n relu(h[row][n] + c128[b][n]) * W2[n]
// W1 staged in LDS as bf16 fragments in contiguous-k order (K = kb*32 + 16*half + e), same as the A fragments.
// __syncthreads() per tile iteration prevents LICM from hoisting all 64
// B-fragment LDS loads out of the loop (which previously caused scratch
// spills); sched_barrier keeps DS loads next to their consuming WMMA.
// ---------------------------------------------------------------------------
__global__ __launch_bounds__(256) void scores_kernel(
        const float* __restrict__ outputs,
        const float* __restrict__ W1,
        const float* __restrict__ W2,
        const float* __restrict__ b2,
        const float* __restrict__ c128,
        float* __restrict__ scores) {
    // Fragment layout: frag (kb, nt), lane l, element e (0..15)
    //   K = kb*32 + (l>>4)*16 + e , N = nt*16 + (l&15)
    __shared__ bf16_t sW[8 * 8 * 32 * 16];   // 64 KB

    const int tid = threadIdx.x;
    // ---- stage + swizzle + convert W1[:256,:128] ----
    for (int i = tid; i < HH * N1; i += 256) {
        const int k = i >> 7;            // 0..255
        const int n = i & 127;           // 0..127
        const int kb = k >> 5, kl = k & 31;
        const int nt = n >> 4, nl = n & 15;
        const int lane = nl + ((kl >> 4) << 4);
        const int e = kl & 15;
        sW[(((kb << 3) + nt) * 32 + lane) * 16 + e] = (bf16_t)W1[(size_t)k * N1 + n];
    }

    const int wave = tid >> 5;
    const int lane = tid & 31;
    const int halfsel = lane >> 4;       // 0 | 1
    const int ml = lane & 15;            // row (A) / col (B,C) within tile
    const int wgTileBase = blockIdx.x * TILES_PER_WG;
    const int b = wgTileBase >> 8;       // 256 tiles per batch; WG stays in one batch

    // hoisted per-lane column constants (col = nt*16 + ml)
    float w2v[8], cv[8];
#pragma unroll
    for (int nt = 0; nt < 8; ++nt) {
        w2v[nt] = W2[nt * 16 + ml];
        cv[nt]  = c128[b * N1 + nt * 16 + ml];
    }
    const float bias2 = b2[0];

    const int ITERS = TILES_PER_WG / WAVES_PER_WG;   // 4
    for (int it = 0; it < ITERS; ++it) {
        // LDS fence: makes the B-fragment loads non-hoistable across
        // iterations (avoids 512-VGPR live range -> scratch spills).
        __syncthreads();

        const int tile = wgTileBase + it * WAVES_PER_WG + wave;
        const int tt = tile & 255;
        const int t0 = tt * 16;
        const float* arow = outputs + ((size_t)b * TT + t0 + ml) * HH;

        if (it + 1 < ITERS) {  // prefetch next tile's row for this lane
            __builtin_prefetch(arow + (size_t)WAVES_PER_WG * 16 * HH, 0, 1);
        }

        v8f acc[8] = {};
#pragma unroll
        for (int kb = 0; kb < 8; ++kb) {
            const int kbase = kb * 32 + halfsel * 16;
            // A fragment (matches B staging): elems 0..15 -> K=kbase..kbase+15
            const float4 x0 = *(const float4*)(arow + kbase);
            const float4 x1 = *(const float4*)(arow + kbase + 4);
            const float4 y0 = *(const float4*)(arow + kbase + 8);
            const float4 y1 = *(const float4*)(arow + kbase + 12);
            v16bf a;
            a[0]  = (bf16_t)x0.x; a[1]  = (bf16_t)x0.y;
            a[2]  = (bf16_t)x0.z; a[3]  = (bf16_t)x0.w;
            a[4]  = (bf16_t)x1.x; a[5]  = (bf16_t)x1.y;
            a[6]  = (bf16_t)x1.z; a[7]  = (bf16_t)x1.w;
            a[8]  = (bf16_t)y0.x; a[9]  = (bf16_t)y0.y;
            a[10] = (bf16_t)y0.z; a[11] = (bf16_t)y0.w;
            a[12] = (bf16_t)y1.x; a[13] = (bf16_t)y1.y;
            a[14] = (bf16_t)y1.z; a[15] = (bf16_t)y1.w;
#pragma unroll
            for (int nt = 0; nt < 8; ++nt) {
                const v16bf bf = *(const v16bf*)&sW[(((kb << 3) + nt) * 32 + lane) * 16];
                acc[nt] = __builtin_amdgcn_wmma_f32_16x16x32_bf16(
                    false, a, false, bf, (short)0, acc[nt], false, false);
            }
            // Keep each k-block's DS loads + WMMAs together; let VALU/VMEM
            // (next A-tile loads, cvt) still be scheduled across.
            SCHED_BARRIER(0x16);
        }

        // ---- fused epilogue: relu + dot with W2 + cross-lane reduce ----
#pragma unroll
        for (int r = 0; r < 8; ++r) {
            float s = 0.f;
#pragma unroll
            for (int nt = 0; nt < 8; ++nt) {
                const float h = acc[nt][r] + cv[nt];
                s = fmaf(fmaxf(h, 0.f), w2v[nt], s);
            }
            // reduce over the 16 lanes of this half (cols 0..15 of each ntile)
            s += __shfl_xor(s, 1, 32);
            s += __shfl_xor(s, 2, 32);
            s += __shfl_xor(s, 4, 32);
            s += __shfl_xor(s, 8, 32);
            if (ml == 0) {
                const int row = r + halfsel * 8;   // C layout: VGPR r -> M = r (+8 hi half)
                scores[(size_t)b * TT + t0 + row] = s + bias2;
            }
        }
    }
}

// ---------------------------------------------------------------------------
// K3: per-batch softmax over T, in place (scores -> normalized weights)
// ---------------------------------------------------------------------------
__global__ void softmax_kernel(float* __restrict__ sc) {
    __shared__ float red[8];
    __shared__ float bcast;
    const int b = blockIdx.x;
    const int tid = threadIdx.x;           // 256 threads
    const int lane = tid & 31, wave = tid >> 5;
    float v[16];
    float m = -3.4e38f;
#pragma unroll
    for (int i = 0; i < 16; ++i) {
        v[i] = sc[(size_t)b * TT + i * 256 + tid];
        m = fmaxf(m, v[i]);
    }
#pragma unroll
    for (int mk = 16; mk >= 1; mk >>= 1) m = fmaxf(m, __shfl_xor(m, mk, 32));
    if (lane == 0) red[wave] = m;
    __syncthreads();
    if (tid < 8) {
        float t = red[tid];
        t = fmaxf(t, __shfl_xor(t, 1, 32));
        t = fmaxf(t, __shfl_xor(t, 2, 32));
        t = fmaxf(t, __shfl_xor(t, 4, 32));
        if (tid == 0) bcast = t;
    }
    __syncthreads();
    m = bcast;
    __syncthreads();

    float s = 0.f;
#pragma unroll
    for (int i = 0; i < 16; ++i) {
        v[i] = __expf(v[i] - m);
        s += v[i];
    }
#pragma unroll
    for (int mk = 16; mk >= 1; mk >>= 1) s += __shfl_xor(s, mk, 32);
    if (lane == 0) red[wave] = s;
    __syncthreads();
    if (tid < 8) {
        float t = red[tid];
        t += __shfl_xor(t, 1, 32);
        t += __shfl_xor(t, 2, 32);
        t += __shfl_xor(t, 4, 32);
        if (tid == 0) bcast = t;
    }
    __syncthreads();
    const float inv = 1.f / bcast;
#pragma unroll
    for (int i = 0; i < 16; ++i)
        sc[(size_t)b * TT + i * 256 + tid] = v[i] * inv;
}

// ---------------------------------------------------------------------------
// K4: split-T weighted pooling partials: partial[b][s][h] over 256 timesteps
// ---------------------------------------------------------------------------
__global__ void pooled_partial_kernel(const float* __restrict__ outputs,
                                      const float* __restrict__ w,
                                      float* __restrict__ partial) {
    __shared__ float sw[256];
    const int b = blockIdx.x, s = blockIdx.y;
    const int h = threadIdx.x;             // 256 threads = H columns
    sw[h] = w[(size_t)b * TT + s * 256 + h];
    __syncthreads();
    const float* base = outputs + ((size_t)b * TT + s * 256) * HH + h;
    float acc = 0.f;
#pragma unroll 4
    for (int t = 0; t < 256; ++t)
        acc = fmaf(sw[t], base[(size_t)t * HH], acc);
    partial[(size_t)((b << 4) + s) * HH + h] = acc;
}

// K5: reduce the 16 split-T partials -> pooled[b][h]
__global__ void pooled_reduce_kernel(const float* __restrict__ partial,
                                     float* __restrict__ pooled) {
    const int b = blockIdx.x;
    const int h = threadIdx.x;             // 256
    float acc = 0.f;
#pragma unroll
    for (int s = 0; s < 16; ++s)
        acc += partial[(size_t)((b << 4) + s) * HH + h];
    pooled[(size_t)b * HH + h] = acc;
}

// ---------------------------------------------------------------------------
// K6: d = relu(pooled @ Wd1 + bd1)     (32x256 @ 256x256 — tiny)
// ---------------------------------------------------------------------------
__global__ void decoder1_kernel(const float* __restrict__ pooled,
                                const float* __restrict__ Wd1,
                                const float* __restrict__ bd1,
                                float* __restrict__ dbuf) {
    __shared__ float sp[HH];
    const int b = blockIdx.x;
    const int n = threadIdx.x;             // 256
    sp[n] = pooled[(size_t)b * HH + n];
    __syncthreads();
    float acc = bd1[n];
    for (int k = 0; k < HH; ++k)
        acc = fmaf(sp[k], Wd1[(size_t)k * HH + n], acc);
    dbuf[(size_t)b * HH + n] = fmaxf(acc, 0.f);
}

// K7: logits = d @ Wd2 + bd2   (32x2)
__global__ void logits_kernel(const float* __restrict__ dbuf,
                              const float* __restrict__ Wd2,
                              const float* __restrict__ bd2,
                              float* __restrict__ out) {
    const int i = threadIdx.x;             // 64 threads
    if (i >= BB * 2) return;
    const int b = i >> 1, o = i & 1;
    float acc = bd2[o];
    for (int k = 0; k < HH; ++k)
        acc = fmaf(dbuf[(size_t)b * HH + k], Wd2[k * 2 + o], acc);
    out[i] = acc;
}

// ---------------------------------------------------------------------------
// Launch
// ---------------------------------------------------------------------------
extern "C" void kernel_launch(void* const* d_in, const int* in_sizes, int n_in,
                              void* d_out, int out_size, void* d_ws, size_t ws_size,
                              hipStream_t stream) {
    const float* outputs = (const float*)d_in[0];
    const float* W1      = (const float*)d_in[1];
    const float* b1      = (const float*)d_in[2];
    const float* W2      = (const float*)d_in[3];
    const float* b2      = (const float*)d_in[4];
    const float* Wd1     = (const float*)d_in[5];
    const float* bd1     = (const float*)d_in[6];
    const float* Wd2     = (const float*)d_in[7];
    const float* bd2     = (const float*)d_in[8];
    float* out = (float*)d_out;

    float* ws      = (float*)d_ws;
    float* c128    = ws;                         // 32*128       = 4096
    float* scores  = c128 + BB * N1;             // 32*4096      = 131072
    float* partial = scores + (size_t)BB * TT;   // 32*16*256    = 131072
    float* pooled  = partial + (size_t)BB * 16 * HH;   // 8192
    float* dbuf    = pooled + (size_t)BB * HH;         // 8192

    c128_kernel<<<BB, 128, 0, stream>>>(outputs, W1, b1, c128);
    scores_kernel<<<NWG, 256, 0, stream>>>(outputs, W1, W2, b2, c128, scores);
    softmax_kernel<<<BB, 256, 0, stream>>>(scores);
    pooled_partial_kernel<<<dim3(BB, 16), 256, 0, stream>>>(outputs, scores, partial);
    pooled_reduce_kernel<<<BB, 256, 0, stream>>>(partial, pooled);
    decoder1_kernel<<<BB, 256, 0, stream>>>(pooled, Wd1, bd1, dbuf);
    logits_kernel<<<1, 64, 0, stream>>>(dbuf, Wd2, bd2, out);
}
